// BertCrf_71253507440916
// MI455X (gfx1250) — matrix operand order repaired and checked by reading, not compiled, not run on hardware
//
#include <hip/hip_runtime.h>
#include <hip/hip_bf16.h>
#include <math.h>

typedef __attribute__((ext_vector_type(16))) __bf16 v16bf;
typedef __attribute__((ext_vector_type(8)))  float  v8f;

constexpr int kB    = 16;
constexpr int kS    = 512;
constexpr int kH    = 768;
constexpr int kNH   = 12;
constexpr int kDH   = 64;
constexpr int kF    = 3072;
constexpr int kL    = 12;
constexpr int kT    = 17;
constexpr int kSM1  = 511;   // S - 1
constexpr int kM    = kB * kS;  // 8192 rows

union FragAB { v16bf v; uint4 u[2]; };

static __device__ inline v8f vzero() {
  v8f z;
  #pragma unroll
  for (int i = 0; i < 8; ++i) z[i] = 0.0f;
  return z;
}

// ---------------------------------------------------------------------------
// fp32 [K x N]  ->  bf16 WT [N x K], 32x32 LDS-tiled (coalesced both sides)
// grid = (N/32, K/32), block = 256 (32x8)
// ---------------------------------------------------------------------------
__launch_bounds__(256)
__global__ void transpose_conv(const float* __restrict__ in, __bf16* __restrict__ out,
                               int K, int N) {
  __shared__ float tile[32][33];
  const int kt = blockIdx.y * 32, nt = blockIdx.x * 32;
  const int tx = threadIdx.x & 31, ty = threadIdx.x >> 5;   // 32 x 8
  #pragma unroll
  for (int i = 0; i < 4; ++i)
    tile[ty + 8 * i][tx] = in[(size_t)(kt + ty + 8 * i) * N + nt + tx];
  __syncthreads();
  #pragma unroll
  for (int i = 0; i < 4; ++i)
    out[(size_t)(nt + ty + 8 * i) * K + kt + tx] = (__bf16)tile[tx][ty + 8 * i];
}

// ---------------------------------------------------------------------------
// Block reduction helper (256 threads)
// ---------------------------------------------------------------------------
static __device__ inline float block_sum256(float v, float* sdata) {
  int tid = threadIdx.x;
  sdata[tid] = v;
  __syncthreads();
  #pragma unroll
  for (int s = 128; s > 0; s >>= 1) {
    if (tid < s) sdata[tid] += sdata[tid + s];
    __syncthreads();
  }
  float r = sdata[0];
  __syncthreads();
  return r;
}

// ---------------------------------------------------------------------------
// Embedding gather + LayerNorm  (one block of 256 per token)
// ---------------------------------------------------------------------------
__launch_bounds__(256)
__global__ void embed_ln(const int* __restrict__ x, const float* __restrict__ wemb,
                         const float* __restrict__ pemb, const float* __restrict__ temb,
                         const float* __restrict__ g, const float* __restrict__ bta,
                         float* __restrict__ hF, __bf16* __restrict__ hB) {
  __shared__ float sdata[256];
  int tok = blockIdx.x;
  int s = tok & (kS - 1);
  int tid = threadIdx.x;
  const float* wrow = wemb + (size_t)x[tok] * kH;
  float v[3];
  float sum = 0.0f;
  #pragma unroll
  for (int i = 0; i < 3; ++i) {
    int c = tid + i * 256;
    float t = wrow[c] + pemb[(size_t)s * kH + c] + temb[c];
    v[i] = t; sum += t;
  }
  float mu = block_sum256(sum, sdata) * (1.0f / kH);
  float sq = 0.0f;
  #pragma unroll
  for (int i = 0; i < 3; ++i) { float d = v[i] - mu; sq += d * d; }
  float var = block_sum256(sq, sdata) * (1.0f / kH);
  float inv = rsqrtf(var + 1e-12f);
  #pragma unroll
  for (int i = 0; i < 3; ++i) {
    int c = tid + i * 256;
    float t = (v[i] - mu) * inv * g[c] + bta[c];
    hF[(size_t)tok * kH + c] = t;
    hB[(size_t)tok * kH + c] = (__bf16)t;
  }
}

// ---------------------------------------------------------------------------
// h = LayerNorm(h + delta)  -> fp32 + bf16
// ---------------------------------------------------------------------------
__launch_bounds__(256)
__global__ void add_ln(float* __restrict__ h, const float* __restrict__ delta,
                       const float* __restrict__ g, const float* __restrict__ bta,
                       __bf16* __restrict__ hB) {
  __shared__ float sdata[256];
  int tok = blockIdx.x;
  int tid = threadIdx.x;
  float v[3];
  float sum = 0.0f;
  #pragma unroll
  for (int i = 0; i < 3; ++i) {
    int c = tid + i * 256;
    float t = h[(size_t)tok * kH + c] + delta[(size_t)tok * kH + c];
    v[i] = t; sum += t;
  }
  float mu = block_sum256(sum, sdata) * (1.0f / kH);
  float sq = 0.0f;
  #pragma unroll
  for (int i = 0; i < 3; ++i) { float d = v[i] - mu; sq += d * d; }
  float var = block_sum256(sq, sdata) * (1.0f / kH);
  float inv = rsqrtf(var + 1e-12f);
  #pragma unroll
  for (int i = 0; i < 3; ++i) {
    int c = tid + i * 256;
    float t = (v[i] - mu) * inv * g[c] + bta[c];
    h[(size_t)tok * kH + c] = t;
    hB[(size_t)tok * kH + c] = (__bf16)t;
  }
}

// ---------------------------------------------------------------------------
// WMMA GEMM: C[M x N] = A[M x K](bf16) * WT[N x K](bf16)^T + bias
// block = 256 threads (8 waves), block tile 128x128, wave tile 32x64
// mode 0: bf16 out   mode 1: GELU -> bf16 out
// mode 2: fp32 out   mode 3: bf16 out scattered as V^T [b,h,d,s]
// ---------------------------------------------------------------------------
__launch_bounds__(256)
__global__ void gemm_bf16(const __bf16* __restrict__ A, const __bf16* __restrict__ WT,
                          const float* __restrict__ bias, int M, int N, int K,
                          int mode, float* __restrict__ outF, __bf16* __restrict__ outB) {
  const int lane = threadIdx.x & 31;
  const int wave = threadIdx.x >> 5;
  const int wm = wave & 3, wn = wave >> 2;
  const int m0 = blockIdx.y * 128 + wm * 32;
  const int n0 = blockIdx.x * 128 + wn * 64;
  const int lm = lane & 15;
  const int kbA = (lane < 16) ? 0 : 8;
  const int kbB = (lane < 16) ? 0 : 8;

  v8f acc[2][4];
  #pragma unroll
  for (int i = 0; i < 2; ++i)
    #pragma unroll
    for (int j = 0; j < 4; ++j) acc[i][j] = vzero();

  for (int k0 = 0; k0 < K; k0 += 32) {
    // Speculative WGP-scope prefetch eight tiles ahead (OOB prefetch is dropped).
    __builtin_prefetch((const void*)(A  + (size_t)(m0 + lm) * K + k0 + 256), 0, 3);
    __builtin_prefetch((const void*)(WT + (size_t)(n0 + lm) * K + k0 + 256), 0, 3);
    FragAB a[2], b[4];
    #pragma unroll
    for (int i = 0; i < 2; ++i) {
      const __bf16* ap = A + (size_t)(m0 + i * 16 + lm) * K + k0 + kbA;
      a[i].u[0] = *(const uint4*)ap;
      a[i].u[1] = *(const uint4*)(ap + 16);
    }
    #pragma unroll
    for (int j = 0; j < 4; ++j) {
      const __bf16* bp = WT + (size_t)(n0 + j * 16 + lm) * K + k0 + kbB;
      b[j].u[0] = *(const uint4*)bp;
      b[j].u[1] = *(const uint4*)(bp + 16);
    }
    #pragma unroll
    for (int i = 0; i < 2; ++i)
      #pragma unroll
      for (int j = 0; j < 4; ++j)
        acc[i][j] = __builtin_amdgcn_wmma_f32_16x16x32_bf16(
            false, a[i].v, false, b[j].v, (short)0, acc[i][j], false, false);
  }

  #pragma unroll
  for (int i = 0; i < 2; ++i) {
    int mbase = m0 + i * 16 + ((lane < 16) ? 0 : 8);
    #pragma unroll
    for (int j = 0; j < 4; ++j) {
      int ncol = n0 + j * 16 + lm;
      float bv = bias[ncol];
      #pragma unroll
      for (int r = 0; r < 8; ++r) {
        int row = mbase + r;
        float v = acc[i][j][r] + bv;
        if (mode == 1) v = 0.5f * v * (1.0f + erff(v * 0.70710678f));
        if (mode == 2) {
          outF[(size_t)row * N + ncol] = v;
        } else if (mode == 3) {
          int bb = row >> 9, ss = row & 511;
          int hd = ncol >> 6, dd = ncol & 63;
          outB[(((size_t)bb * kNH + hd) * kDH + dd) * kS + ss] = (__bf16)v;
        } else {
          outB[(size_t)row * N + ncol] = (__bf16)v;
        }
      }
    }
  }
}

// ---------------------------------------------------------------------------
// Fused flash attention: one wave per 16 query rows, online softmax,
// P tile restaged through LDS, 8 WMMAs per 32-key chunk.
// grid = (B*NH, S/128), block = 256 (8 waves)
// ---------------------------------------------------------------------------
__launch_bounds__(256)
__global__ void attention(const __bf16* __restrict__ Q, const __bf16* __restrict__ Kb,
                          const __bf16* __restrict__ VT, const int* __restrict__ mask,
                          __bf16* __restrict__ ctx) {
  __shared__ __bf16 Plds[8][16 * 32];
  const int lane = threadIdx.x & 31;
  const int wave = threadIdx.x >> 5;
  const int bh = blockIdx.x;
  const int b = bh / kNH, hd = bh % kNH;
  const int q0 = blockIdx.y * 128 + wave * 16;
  const int lm = lane & 15;
  const int kbA = (lane < 16) ? 0 : 8;
  const int kbB = (lane < 16) ? 0 : 8;
  const float scale = 0.125f;  // 1/sqrt(64)

  FragAB qa[2];
  #pragma unroll
  for (int i = 0; i < 2; ++i) {
    const __bf16* qp = Q + (size_t)(b * kS + q0 + lm) * kH + hd * kDH + i * 32 + kbA;
    qa[i].u[0] = *(const uint4*)qp;
    qa[i].u[1] = *(const uint4*)(qp + 16);
  }

  float run_m[8], run_l[8];
  v8f O[4];
  #pragma unroll
  for (int r = 0; r < 8; ++r) { run_m[r] = -3.0e38f; run_l[r] = 0.0f; }
  #pragma unroll
  for (int j = 0; j < 4; ++j) O[j] = vzero();

  for (int kc = 0; kc < kS; kc += 32) {
    v8f s0 = vzero(), s1 = vzero();
    #pragma unroll
    for (int i = 0; i < 2; ++i) {
      FragAB kb0, kb1;
      const __bf16* kp0 = Kb + (size_t)(b * kS + kc + lm) * kH + hd * kDH + i * 32 + kbB;
      const __bf16* kp1 = kp0 + (size_t)16 * kH;
      kb0.u[0] = *(const uint4*)kp0;  kb0.u[1] = *(const uint4*)(kp0 + 16);
      kb1.u[0] = *(const uint4*)kp1;  kb1.u[1] = *(const uint4*)(kp1 + 16);
      s0 = __builtin_amdgcn_wmma_f32_16x16x32_bf16(false, qa[i].v, false, kb0.v, (short)0, s0, false, false);
      s1 = __builtin_amdgcn_wmma_f32_16x16x32_bf16(false, qa[i].v, false, kb1.v, (short)0, s1, false, false);
    }

    float bias0 = (mask[b * kS + kc + lm]      != 0) ? 0.0f : -10000.0f;
    float bias1 = (mask[b * kS + kc + 16 + lm] != 0) ? 0.0f : -10000.0f;

    float p0[8], p1[8];
    #pragma unroll
    for (int r = 0; r < 8; ++r) {
      float a0 = s0[r] * scale + bias0;
      float a1 = s1[r] * scale + bias1;
      float mx = fmaxf(a0, a1);
      #pragma unroll
      for (int d = 1; d < 16; d <<= 1) mx = fmaxf(mx, __shfl_xor(mx, d, 16));
      float nm = fmaxf(run_m[r], mx);
      float sc = __expf(run_m[r] - nm);
      float e0 = __expf(a0 - nm), e1 = __expf(a1 - nm);
      float rs = e0 + e1;
      #pragma unroll
      for (int d = 1; d < 16; d <<= 1) rs += __shfl_xor(rs, d, 16);
      run_l[r] = run_l[r] * sc + rs;
      run_m[r] = nm;
      #pragma unroll
      for (int j = 0; j < 4; ++j) O[j][r] *= sc;
      p0[r] = e0; p1[r] = e1;
    }

    // stage P tile (16x32) into LDS, reload as A fragment
    int prow = (lane < 16) ? 0 : 8;
    #pragma unroll
    for (int r = 0; r < 8; ++r) {
      Plds[wave][(prow + r) * 32 + lm]      = (__bf16)p0[r];
      Plds[wave][(prow + r) * 32 + 16 + lm] = (__bf16)p1[r];
    }
    __syncthreads();
    FragAB pf;
    const __bf16* pp = &Plds[wave][lm * 32 + kbA];
    pf.u[0] = *(const uint4*)pp;
    pf.u[1] = *(const uint4*)(pp + 16);
    __syncthreads();

    #pragma unroll
    for (int j = 0; j < 4; ++j) {
      FragAB vb;
      const __bf16* vp = VT + ((size_t)(b * kNH + hd) * kDH + j * 16 + lm) * kS + kc + kbB;
      vb.u[0] = *(const uint4*)vp;
      vb.u[1] = *(const uint4*)(vp + 16);
      O[j] = __builtin_amdgcn_wmma_f32_16x16x32_bf16(false, pf.v, false, vb.v, (short)0, O[j], false, false);
    }
  }

  int rbase = q0 + ((lane < 16) ? 0 : 8);
  #pragma unroll
  for (int r = 0; r < 8; ++r) {
    float inv = 1.0f / run_l[r];
    int row = rbase + r;
    #pragma unroll
    for (int j = 0; j < 4; ++j) {
      int col = hd * kDH + j * 16 + lm;
      ctx[(size_t)(b * kS + row) * kH + col] = (__bf16)(O[j][r] * inv);
    }
  }
}

// ---------------------------------------------------------------------------
// logits[b, t, tag] = h[b, t+1, :] @ out_W + out_b  (one block per token)
// h row staged via CDNA5 async global->LDS DMA
// ---------------------------------------------------------------------------
__launch_bounds__(256)
__global__ void logits_k(const float* __restrict__ h, const float* __restrict__ outW,
                         const float* __restrict__ outBias, float* __restrict__ logits) {
  __shared__ float sh[kH];
  __shared__ float acc[kT];
  int idx = blockIdx.x;          // b * 511 + t
  int b = idx / kSM1, t = idx % kSM1;
  const float* hrow = h + (size_t)(b * kS + t + 1) * kH;
  int tid = threadIdx.x;
  if (tid < kH / 4) {  // 192 lanes copy 16B each: 768 floats
    unsigned lds_addr = (unsigned)(size_t)&sh[tid * 4];
    const float* gp = hrow + tid * 4;
    asm volatile("global_load_async_to_lds_b128 %0, %1, off"
                 :: "v"(lds_addr), "v"(gp) : "memory");
  }
  asm volatile("s_wait_asynccnt 0x0" ::: "memory");
  if (tid < kT) acc[tid] = outBias[tid];
  __syncthreads();
  if (tid < 255) {
    int tag = tid / 15, p = tid % 15;
    float s = 0.0f;
    for (int c = p; c < kH; c += 15) s += sh[c] * outW[(size_t)c * kT + tag];
    atomicAdd(&acc[tag], s);
  }
  __syncthreads();
  if (tid < kT) logits[(size_t)idx * kT + tid] = acc[tid];
}

// ---------------------------------------------------------------------------
// CRF forward: one block, 16 batches x 32 lanes; alpha recursion + numerator
// ---------------------------------------------------------------------------
__launch_bounds__(512)
__global__ void crf_k(const float* __restrict__ logits, const float* __restrict__ trans,
                      const int* __restrict__ y, const int* __restrict__ mask,
                      float* __restrict__ out) {
  __shared__ float T[kT * kT];
  __shared__ float bres[kB];
  int tid = threadIdx.x;
  for (int i = tid; i < kT * kT; i += 512) T[i] = trans[i];
  __syncthreads();
  int b = tid >> 5, lane = tid & 31;
  const float NEG = -3.0e38f;
  const float* lg = logits + (size_t)b * kSM1 * kT;
  float alpha = (lane < kT) ? lg[lane] : NEG;
  for (int t = 1; t < kSM1; ++t) {
    float lt = (lane < kT) ? lg[(size_t)t * kT + lane] : 0.0f;
    float mx = NEG;
    for (int i = 0; i < kT; ++i) {
      float ai = __shfl(alpha, i, 32);
      float v = ai + ((lane < kT) ? T[i * kT + lane] : 0.0f);
      mx = fmaxf(mx, v);
    }
    float s = 0.0f;
    for (int i = 0; i < kT; ++i) {
      float ai = __shfl(alpha, i, 32);
      float v = ai + ((lane < kT) ? T[i * kT + lane] : 0.0f);
      s += __expf(v - mx);
    }
    float nw = mx + __logf(s) + lt;
    int mt = mask[b * kS + 1 + t];
    if (lane < kT) alpha = (mt > 0) ? nw : alpha;
  }
  // denom = logsumexp over tags
  float mx = alpha;
  #pragma unroll
  for (int d = 1; d < 32; d <<= 1) mx = fmaxf(mx, __shfl_xor(mx, d, 32));
  float s = (lane < kT) ? __expf(alpha - mx) : 0.0f;
  #pragma unroll
  for (int d = 1; d < 32; d <<= 1) s += __shfl_xor(s, d, 32);
  float denom = mx + __logf(s);
  // numerator
  const int* yb = y + b * kSM1;
  float part = 0.0f;
  for (int t = lane; t < kSM1 - 1; t += 32) {
    float cm_t  = (mask[b * kS + 1 + t] > 0) ? 1.0f : 0.0f;
    float cm_t1 = (mask[b * kS + 2 + t] > 0) ? 1.0f : 0.0f;
    part += lg[(size_t)t * kT + yb[t]] * cm_t;
    part += T[yb[t] * kT + yb[t + 1]] * cm_t1;
  }
  #pragma unroll
  for (int d = 1; d < 32; d <<= 1) part += __shfl_xor(part, d, 32);
  if (lane == 0) {
    int cnt = 0;
    for (int t = 0; t < kSM1; ++t) cnt += (mask[b * kS + 1 + t] > 0) ? 1 : 0;
    int last_tag = yb[cnt - 1];
    float cm_last = (mask[b * kS + kS - 1] > 0) ? 1.0f : 0.0f;
    float num = part + lg[(size_t)(kSM1 - 1) * kT + last_tag] * cm_last;
    bres[b] = num - denom;
  }
  __syncthreads();
  if (tid == 0) {
    float tot = 0.0f;
    for (int i = 0; i < kB; ++i) tot += bres[i];
    out[0] = tot;
  }
}

// ---------------------------------------------------------------------------
extern "C" void kernel_launch(void* const* d_in, const int* in_sizes, int n_in,
                              void* d_out, int out_size, void* d_ws, size_t ws_size,
                              hipStream_t stream) {
  const int*   x     = (const int*)d_in[0];
  const int*   y     = (const int*)d_in[1];
  const int*   mask  = (const int*)d_in[2];
  const float* wemb  = (const float*)d_in[3];
  const float* pemb  = (const float*)d_in[4];
  const float* temb  = (const float*)d_in[5];
  const float* elng  = (const float*)d_in[6];
  const float* elnb  = (const float*)d_in[7];
  const float* Wq    = (const float*)d_in[8];
  const float* bq    = (const float*)d_in[9];
  const float* Wk    = (const float*)d_in[10];
  const float* bk    = (const float*)d_in[11];
  const float* Wv    = (const float*)d_in[12];
  const float* bv    = (const float*)d_in[13];
  const float* Wo    = (const float*)d_in[14];
  const float* bo    = (const float*)d_in[15];
  const float* ln1g  = (const float*)d_in[16];
  const float* ln1b  = (const float*)d_in[17];
  const float* Wff1  = (const float*)d_in[18];
  const float* bff1  = (const float*)d_in[19];
  const float* Wff2  = (const float*)d_in[20];
  const float* bff2  = (const float*)d_in[21];
  const float* ln2g  = (const float*)d_in[22];
  const float* ln2b  = (const float*)d_in[23];
  const float* outW  = (const float*)d_in[24];
  const float* outBi = (const float*)d_in[25];
  const float* transitions = (const float*)d_in[26];

  char* w = (char*)d_ws;
  size_t off = 0;
  auto alloc = [&](size_t bytes) { size_t o = off; off += (bytes + 255) & ~(size_t)255; return o; };
  const size_t MH4 = (size_t)kM * kH * 4;
  const size_t MH2 = (size_t)kM * kH * 2;
  const size_t MF2 = (size_t)kM * kF * 2;

  float*  hF    = (float*) (w + alloc(MH4));
  __bf16* hB    = (__bf16*)(w + alloc(MH2));
  float*  tmpF  = (float*) (w + alloc(MH4));
  __bf16* qB    = (__bf16*)(w + alloc(MH2));
  __bf16* kB_   = (__bf16*)(w + alloc(MH2));
  __bf16* vT    = (__bf16*)(w + alloc(MH2));
  __bf16* ctxB  = (__bf16*)(w + alloc(MH2));
  __bf16* ffB   = (__bf16*)(w + alloc(MF2));
  __bf16* wtq   = (__bf16*)(w + alloc((size_t)kH * kH * 2));
  __bf16* wtk   = (__bf16*)(w + alloc((size_t)kH * kH * 2));
  __bf16* wtv   = (__bf16*)(w + alloc((size_t)kH * kH * 2));
  __bf16* wto   = (__bf16*)(w + alloc((size_t)kH * kH * 2));
  __bf16* wt1   = (__bf16*)(w + alloc((size_t)kH * kF * 2));
  __bf16* wt2   = (__bf16*)(w + alloc((size_t)kF * kH * 2));
  float*  logit = (float*) (w + alloc((size_t)kB * kSM1 * kT * 4));

  const int HH = kH * kH, HF = kH * kF;
  const dim3 gHH(kH / 32, kH / 32);        // K=768,  N=768
  const dim3 gFF1(kF / 32, kH / 32);       // K=768,  N=3072
  const dim3 gFF2(kH / 32, kF / 32);       // K=3072, N=768
  const dim3 gemmH(kH / 128, kM / 128);    // N=768
  const dim3 gemmF(kF / 128, kM / 128);    // N=3072
  const dim3 attG(kB * kNH, kS / 128);

  embed_ln<<<kM, 256, 0, stream>>>(x, wemb, pemb, temb, elng, elnb, hF, hB);

  for (int l = 0; l < kL; ++l) {
    transpose_conv<<<gHH, 256, 0, stream>>>(Wq + (size_t)l * HH, wtq, kH, kH);
    transpose_conv<<<gHH, 256, 0, stream>>>(Wk + (size_t)l * HH, wtk, kH, kH);
    transpose_conv<<<gHH, 256, 0, stream>>>(Wv + (size_t)l * HH, wtv, kH, kH);
    transpose_conv<<<gHH, 256, 0, stream>>>(Wo + (size_t)l * HH, wto, kH, kH);
    transpose_conv<<<gFF1, 256, 0, stream>>>(Wff1 + (size_t)l * HF, wt1, kH, kF);
    transpose_conv<<<gFF2, 256, 0, stream>>>(Wff2 + (size_t)l * HF, wt2, kF, kH);

    gemm_bf16<<<gemmH, 256, 0, stream>>>(hB, wtq, bq + l * kH, kM, kH, kH, 0, nullptr, qB);
    gemm_bf16<<<gemmH, 256, 0, stream>>>(hB, wtk, bk + l * kH, kM, kH, kH, 0, nullptr, kB_);
    gemm_bf16<<<gemmH, 256, 0, stream>>>(hB, wtv, bv + l * kH, kM, kH, kH, 3, nullptr, vT);

    attention<<<attG, 256, 0, stream>>>(qB, kB_, vT, mask, ctxB);

    gemm_bf16<<<gemmH, 256, 0, stream>>>(ctxB, wto, bo + l * kH, kM, kH, kH, 2, tmpF, nullptr);
    add_ln<<<kM, 256, 0, stream>>>(hF, tmpF, ln1g + l * kH, ln1b + l * kH, hB);

    gemm_bf16<<<gemmF, 256, 0, stream>>>(hB, wt1, bff1 + l * kF, kM, kF, kH, 1, nullptr, ffB);
    gemm_bf16<<<gemmH, 256, 0, stream>>>(ffB, wt2, bff2 + l * kH, kM, kH, kF, 2, tmpF, nullptr);
    add_ln<<<kM, 256, 0, stream>>>(hF, tmpF, ln2g + l * kH, ln2b + l * kH, hB);
  }

  logits_k<<<kB * kSM1, 256, 0, stream>>>(hF, outW, outBi, logit);
  crf_k<<<1, 512, 0, stream>>>(logit, transitions, y, mask, (float*)d_out);

  (void)in_sizes; (void)n_in; (void)out_size; (void)ws_size;
}
